// DynamicLayer_27376121544752
// MI455X (gfx1250) — matrix operand order repaired and checked by reading, not compiled, not run on hardware
//
#include <hip/hip_runtime.h>

typedef __attribute__((ext_vector_type(16))) _Float16 v16h;
typedef __attribute__((ext_vector_type(8)))  _Float16 v8h;
typedef __attribute__((ext_vector_type(8)))  float    v8f;
typedef __attribute__((ext_vector_type(4)))  float    f32x4;

// THRESHOLD(=1.0) * D(=128): mean|x| > 1  <=>  sum|x| > 128 (exact: /128 is a pow2)
#define THRESH_SUM 128.0f

__device__ __forceinline__ v8f wmma_f16(v16h a, v16h b, v8f c) {
    // 8 args: (neg_a, A, neg_b, B, c_mod, C, reuse_a, reuse_b)
    return __builtin_amdgcn_wmma_f32_16x16x32_f16(false, a, false, b, (short)0, c,
                                                  false, false);
}

constexpr int D_    = 128;
constexpr int U1_   = 64;
constexpr int U2_   = 32;
constexpr int NWAVE = 8;   // waves per block (wave32)

__global__ __launch_bounds__(256)
void moe_fused(const float* __restrict__ x,
               const float* __restrict__ W1, const float* __restrict__ b1,
               const float* __restrict__ W2, const float* __restrict__ b2,
               const float* __restrict__ Wf, const float* __restrict__ bf,
               float* __restrict__ out, int nrows)
{
    __shared__ _Float16 sW1[U1_ * D_];        // 16 KB
    __shared__ _Float16 sW2[U2_ * D_];        //  8 KB
    __shared__ _Float16 sWf[D_  * U1_];       // 16 KB
    __shared__ _Float16 sBr[NWAVE][16 * U1_]; // 16 KB (per-wave branch scratch)

    const int tid = threadIdx.x;
    // Stage weights once per block as f16 (they are reused by every tile).
    for (int i = tid; i < U1_ * D_;  i += 256) sW1[i] = (_Float16)W1[i];
    for (int i = tid; i < U2_ * D_;  i += 256) sW2[i] = (_Float16)W2[i];
    for (int i = tid; i < D_  * U1_; i += 256) sWf[i] = (_Float16)Wf[i];
    __syncthreads();

    const int wave = tid >> 5;
    const int lane = tid & 31;
    const int half = lane >> 4;   // which 16-lane half of the wave
    const int l16  = lane & 15;

    const int row0 = (blockIdx.x * NWAVE + wave) * 16;
    if (row0 >= nrows) return;    // wave-uniform guard (EXEC stays all-ones)

    // ---- Load x tile straight into WMMA A-layout (f32 -> f16), |x| sums free ----
    // 16-bit A 16x32: lane half h holds K = kk*32 + 16h + [0..15] contiguous
    const float* xrow = x + (size_t)(row0 + l16) * D_;
    v16h  afrag[4];
    float asum = 0.0f;
#pragma unroll
    for (int kk = 0; kk < 4; ++kk) {
        const int kb = kk * 32 + 16 * half;
        f32x4 q0 = __builtin_nontemporal_load((const f32x4*)(xrow + kb));
        f32x4 q1 = __builtin_nontemporal_load((const f32x4*)(xrow + kb + 4));
        f32x4 q2 = __builtin_nontemporal_load((const f32x4*)(xrow + kb + 8));
        f32x4 q3 = __builtin_nontemporal_load((const f32x4*)(xrow + kb + 12));
        float v[16] = {q0.x,q0.y,q0.z,q0.w, q1.x,q1.y,q1.z,q1.w,
                       q2.x,q2.y,q2.z,q2.w, q3.x,q3.y,q3.z,q3.w};
#pragma unroll
        for (int i = 0; i < 16; ++i) {
            afrag[kk][i] = (_Float16)v[i];
            asum += fabsf(v[i]);
        }
    }
    // lane L holds half of row L%16; combine halves, then every lane has full row sum
    asum += __shfl_xor(asum, 16, 32);
    // rsel[r] = row-sum of row (r + 8*half)  (C/D layout row index for VGPR r)
    float rsel[8];
#pragma unroll
    for (int r = 0; r < 8; ++r)
        rsel[r] = __shfl(asum, r + 8 * half, 32);

    // ---- GEMM1 (x @ W1^T) and GEMM2 (x @ W2^T), bias folded into accumulators ----
    v8f acc1[4], acc2[2];
#pragma unroll
    for (int nt = 0; nt < 4; ++nt) {
        const float bv = b1[nt * 16 + l16];       // D-layout: N = l16
#pragma unroll
        for (int i = 0; i < 8; ++i) acc1[nt][i] = bv;
    }
#pragma unroll
    for (int nt = 0; nt < 2; ++nt) {
        const float bv = b2[nt * 16 + l16];
#pragma unroll
        for (int i = 0; i < 8; ++i) acc2[nt][i] = bv;
    }
    // 16-bit B 32x16: lane half h holds column N=l16, K = kk*32 + 16h + [0..15] contiguous
#pragma unroll
    for (int kk = 0; kk < 4; ++kk) {
        const int kb = kk * 32 + 16 * half;
#pragma unroll
        for (int nt = 0; nt < 4; ++nt) {
            v16h bfr = *(const v16h*)(sW1 + (nt * 16 + l16) * D_ + kb);
            acc1[nt] = wmma_f16(afrag[kk], bfr, acc1[nt]);
        }
#pragma unroll
        for (int nt = 0; nt < 2; ++nt) {
            v16h bfr = *(const v16h*)(sW2 + (nt * 16 + l16) * D_ + kb);
            acc2[nt] = wmma_f16(afrag[kk], bfr, acc2[nt]);
        }
    }

    // ---- Per-row select + relu; write 16x64 branch tile row-major to LDS ----
    _Float16* br = sBr[wave];                      // private to this wave
#pragma unroll
    for (int nt = 0; nt < 4; ++nt) {
#pragma unroll
        for (int r = 0; r < 8; ++r) {
            const float o1v = acc1[nt][r];
            const float o2v = (nt < 2) ? acc2[nt][r] : 0.0f;  // zero-pad cols 32..63
            float vv = (rsel[r] > THRESH_SUM) ? o1v : o2v;
            vv = fmaxf(vv, 0.0f);
            br[(r + 8 * half) * U1_ + nt * 16 + l16] = (_Float16)vv;
        }
    }
    // same-wave DS ops are in-order; region is wave-private, no barrier needed

    // ---- Reload branch tile in A-layout for GEMM3 (K = 64 -> 2 k-steps) ----
    v16h a3[2];
#pragma unroll
    for (int kk = 0; kk < 2; ++kk) {
        const int kb = kk * 32 + 16 * half;
        v8h lo = *(const v8h*)(br + l16 * U1_ + kb);
        v8h hi = *(const v8h*)(br + l16 * U1_ + kb + 8);
#pragma unroll
        for (int i = 0; i < 8; ++i) { a3[kk][i] = lo[i]; a3[kk][8 + i] = hi[i]; }
    }

    // ---- GEMM3 (branch @ Wf^T) + bf, streamed stores ----
#pragma unroll
    for (int nt = 0; nt < 8; ++nt) {
        const float bv = bf[nt * 16 + l16];
        v8f acc;
#pragma unroll
        for (int i = 0; i < 8; ++i) acc[i] = bv;
#pragma unroll
        for (int kk = 0; kk < 2; ++kk) {
            v16h bfr = *(const v16h*)(sWf + (nt * 16 + l16) * U1_ + kk * 32 + 16 * half);
            acc = wmma_f16(a3[kk], bfr, acc);
        }
#pragma unroll
        for (int r = 0; r < 8; ++r) {
            __builtin_nontemporal_store(
                acc[r], out + (size_t)(row0 + r + 8 * half) * D_ + nt * 16 + l16);
        }
    }
}

extern "C" void kernel_launch(void* const* d_in, const int* in_sizes, int n_in,
                              void* d_out, int out_size, void* d_ws, size_t ws_size,
                              hipStream_t stream) {
    const float* x  = (const float*)d_in[0];
    const float* W1 = (const float*)d_in[1];
    const float* b1 = (const float*)d_in[2];
    const float* W2 = (const float*)d_in[3];
    const float* b2 = (const float*)d_in[4];
    const float* Wf = (const float*)d_in[5];
    const float* bf = (const float*)d_in[6];
    float* out = (float*)d_out;

    const int nrows  = in_sizes[0] / D_;               // B = 262144
    const int blocks = (nrows + (NWAVE * 16) - 1) / (NWAVE * 16);
    moe_fused<<<blocks, 256, 0, stream>>>(x, W1, b1, W2, b2, Wf, bf, out, nrows);
}
